// MultiHeadAttentionRoPE_67723044323464
// MI455X (gfx1250) — matrix operand order repaired and checked by reading, not compiled, not run on hardware
//
#include <hip/hip_runtime.h>
#include <hip/hip_bf16.h>

typedef _Float16 half_t;
typedef __attribute__((ext_vector_type(16))) _Float16 v16h;
typedef __attribute__((ext_vector_type(8)))  _Float16 v8h;
typedef __attribute__((ext_vector_type(8)))  float    v8f;

#define D_MODEL 1024
#define SEQ_T   2048
#define NB      2
#define NH      16
#define DK      64
#define M_TOT   4096   // NB*SEQ_T

// ---------------------------------------------------------------------------
// WMMA fragment helpers (CDNA5 v_wmma_f32_16x16x32_f16, wave32).
// A: 16x32 (MxK) f16.  Lane l: row = l&15, kb = (l>>4)*16; holds K kb..kb+15.
// B: 32x16 (KxN) f16.  Lane l: col n = l&15, kb = (l>>4)*16; holds K kb..kb+15.
// C/D: 16x16 f32.      Lane l: n = l&15; VGPR v -> M = v + 8*(l>>4).
// ---------------------------------------------------------------------------
static __device__ __forceinline__ v16h load_a_frag(const half_t* base, int ld) {
    const int l   = threadIdx.x & 31;
    const int row = l & 15;
    const int kb  = (l >> 4) << 4;
    const half_t* p = base + row * ld + kb;
    return *(const v16h*)(p);
}

// B element [k][n] lives at base[n*ld + k] (column n contiguous in k).
static __device__ __forceinline__ v16h load_b_frag(const half_t* base, int ld) {
    const int l  = threadIdx.x & 31;
    const int n  = l & 15;
    const int kb = (l >> 4) << 4;
    return *(const v16h*)(base + n * ld + kb);
}

static __device__ __forceinline__ v8f wmma16(v16h a, v16h b, v8f c) {
    return __builtin_amdgcn_wmma_f32_16x16x32_f16(false, a, false, b,
                                                  (short)0, c, false, false);
}

// ---------------------------------------------------------------------------
// Software-pipelined 32x64 GEMM tile with ping-pong fragment buffers:
// acc[2][4] += X[m0:m0+32, :] * W[n0:n0+64, :]^T   (K = D_MODEL).
// Two buffer sets alternate as load/compute targets -> no register copies,
// no WMMA->VALU WAR hazard NOPs, loads of one set overlap WMMAs on the other.
// ---------------------------------------------------------------------------
static __device__ __forceinline__ void ld_frags(
    const half_t* __restrict__ X, const half_t* __restrict__ W,
    int m0, int n0, int kk, v16h a[2], v16h b[4])
{
    a[0] = load_a_frag(X + (size_t)m0 * D_MODEL + kk, D_MODEL);
    a[1] = load_a_frag(X + (size_t)(m0 + 16) * D_MODEL + kk, D_MODEL);
#pragma unroll
    for (int j = 0; j < 4; ++j)
        b[j] = load_b_frag(W + (size_t)(n0 + j * 16) * D_MODEL + kk, D_MODEL);
}

static __device__ __forceinline__ void do_wmma8(v16h a[2], v16h b[4],
                                                v8f acc[2][4])
{
#pragma unroll
    for (int mi = 0; mi < 2; ++mi)
#pragma unroll
        for (int j = 0; j < 4; ++j)
            acc[mi][j] = wmma16(a[mi], b[j], acc[mi][j]);
}

static __device__ __forceinline__ void mm32x64(
    const half_t* __restrict__ X, const half_t* __restrict__ W,
    int m0, int n0, v8f acc[2][4])
{
    v16h a0[2], b0[4], a1[2], b1[4];
    ld_frags(X, W, m0, n0, 0, a0, b0);                 // chunk 0 -> buf0
    for (int k0 = 0; k0 < D_MODEL - 64; k0 += 64) {    // 15 iters (pairs 0..14)
        ld_frags(X, W, m0, n0, k0 + 32, a1, b1);       // odd chunk -> buf1
        do_wmma8(a0, b0, acc);
        ld_frags(X, W, m0, n0, k0 + 64, a0, b0);       // next even -> buf0
        do_wmma8(a1, b1, acc);
    }
    ld_frags(X, W, m0, n0, D_MODEL - 32, a1, b1);      // chunk 31 -> buf1
    do_wmma8(a0, b0, acc);                             // chunk 30
    do_wmma8(a1, b1, acc);                             // chunk 31
}

// ---------------------------------------------------------------------------
// Kernel 0: fp32 -> f16 convert (bandwidth-trivial at 23.3 TB/s).
// ---------------------------------------------------------------------------
__global__ void cvt_f32_f16_kernel(const float* __restrict__ s,
                                   half_t* __restrict__ d, int n) {
    int i = blockIdx.x * blockDim.x + threadIdx.x;
    if (i < n) d[i] = (half_t)s[i];
}

// ---------------------------------------------------------------------------
// Kernel 1: fused Q/K/V projection + RoPE.
// grid = (32, NH, 3), block = 128 (4 waves). Wave -> 32 tokens x 1 head.
// which = blockIdx.z: 0=Q(+RoPE), 1=K(+RoPE), 2=V (stored transposed [bh][d][t]).
// ---------------------------------------------------------------------------
__global__ __launch_bounds__(128) void qkv_rope_kernel(
    const half_t* __restrict__ q16, const half_t* __restrict__ k16,
    const half_t* __restrict__ v16, const half_t* __restrict__ Wq16,
    const half_t* __restrict__ Wk16, const half_t* __restrict__ Wv16,
    half_t* __restrict__ Qh, half_t* __restrict__ Kh, half_t* __restrict__ Vt)
{
    const int which = blockIdx.z;
    const half_t* X = (which == 0) ? q16 : (which == 1) ? k16 : v16;
    const half_t* W = (which == 0) ? Wq16 : (which == 1) ? Wk16 : Wv16;

    const int wid  = threadIdx.x >> 5;
    const int l    = threadIdx.x & 31;
    const int hi   = l >> 4;
    const int mt   = blockIdx.x * 4 + wid;      // 0..127
    const int head = blockIdx.y;
    const int m0   = mt * 32;
    const int n0   = head * DK;

    v8f acc[2][4] = {};
    mm32x64(X, W, m0, n0, acc);

    const int b  = m0 >> 11;                    // token row -> batch
    const int bh = b * NH + head;

#pragma unroll
    for (int mi = 0; mi < 2; ++mi) {
        const int tbase = ((m0 + mi * 16) & 2047) + 8 * hi;
        if (which < 2) {
            // RoPE on f32 accumulators: within-head dim d = j*16 + (l&15);
            // rotation pair (2i, 2i+1) == adjacent lanes (lane^1) of same frag.
            half_t* dst = ((which == 0) ? Qh : Kh) + (size_t)bh * (SEQ_T * DK);
#pragma unroll
            for (int j = 0; j < 4; ++j) {
                const int d  = j * 16 + (l & 15);
                const int fi = d >> 1;
                const float invf = __expf(-(float)fi * 0.28782314f); // ln(1e4)/32
#pragma unroll
                for (int v = 0; v < 8; ++v) {
                    float t  = (float)(tbase + v);
                    float fr = t * invf;
                    float c  = __cosf(fr);
                    float s  = __sinf(fr);
                    float x  = acc[mi][j][v];
                    float p  = __shfl_xor(x, 1, 32);
                    float r  = (d & 1) ? (p * s + x * c) : (x * c - p * s);
                    dst[(size_t)(tbase + v) * DK + d] = (half_t)r;
                }
            }
        } else {
            // V: store transposed [bh][d][t] so PV B-fragments are contiguous.
            half_t* dst = Vt + (size_t)bh * (SEQ_T * DK);
#pragma unroll
            for (int j = 0; j < 4; ++j) {
                const int d = j * 16 + (l & 15);
                v8h pk;
#pragma unroll
                for (int v = 0; v < 8; ++v) pk[v] = (half_t)acc[mi][j][v];
                *(v8h*)(dst + (size_t)d * SEQ_T + tbase) = pk;
            }
        }
    }
}

// ---------------------------------------------------------------------------
// Kernel 2: flash attention, one 16-row Q tile per wave, 32-key blocks.
// grid = (32, NB*NH), block = 128 (4 waves).
// ---------------------------------------------------------------------------
__global__ __launch_bounds__(128) void attn_kernel(
    const half_t* __restrict__ Qh, const half_t* __restrict__ Kh,
    const half_t* __restrict__ Vt, half_t* __restrict__ AO)
{
    __shared__ __align__(16) half_t pbuf[4][16 * 32];

    const int wid = threadIdx.x >> 5;
    const int l   = threadIdx.x & 31;
    const int hi  = l >> 4;
    const int bh  = blockIdx.y;
    const int q0  = (blockIdx.x * 4 + wid) * 16;

    const half_t* Qb = Qh + (size_t)bh * (SEQ_T * DK);
    const half_t* Kb = Kh + (size_t)bh * (SEQ_T * DK);
    const half_t* Vb = Vt + (size_t)bh * (SEQ_T * DK);
    half_t* P = &pbuf[wid][0];

    v16h aQ0 = load_a_frag(Qb + (size_t)q0 * DK + 0, DK);
    v16h aQ1 = load_a_frag(Qb + (size_t)q0 * DK + 32, DK);

    v8f o0 = {}, o1 = {}, o2 = {}, o3 = {};
    float mrow[8], lrow[8];
#pragma unroll
    for (int v = 0; v < 8; ++v) { mrow[v] = -1e30f; lrow[v] = 0.0f; }

    for (int kt = 0; kt < SEQ_T; kt += 32) {
        if (kt + 32 < SEQ_T)
            __builtin_prefetch(Kb + (size_t)(kt + 32) * DK + l * 4, 0, 0);

        // --- hoist all K fragments, then 4 back-to-back WMMAs ---
        v16h bK00 = load_b_frag(Kb + (size_t)kt * DK + 0,  DK);
        v16h bK01 = load_b_frag(Kb + (size_t)kt * DK + 32, DK);
        v16h bK10 = load_b_frag(Kb + (size_t)(kt + 16) * DK + 0,  DK);
        v16h bK11 = load_b_frag(Kb + (size_t)(kt + 16) * DK + 32, DK);
        // --- issue V fragment loads now; latency hides under the softmax ---
        v16h bV0 = load_b_frag(Vb + (size_t)0  * SEQ_T + kt, SEQ_T);
        v16h bV1 = load_b_frag(Vb + (size_t)16 * SEQ_T + kt, SEQ_T);
        v16h bV2 = load_b_frag(Vb + (size_t)32 * SEQ_T + kt, SEQ_T);
        v16h bV3 = load_b_frag(Vb + (size_t)48 * SEQ_T + kt, SEQ_T);

        v8f s0 = {}, s1 = {};
        s0 = wmma16(aQ0, bK00, s0);
        s0 = wmma16(aQ1, bK01, s0);
        s1 = wmma16(aQ0, bK10, s1);
        s1 = wmma16(aQ1, bK11, s1);

        const float sc = 0.125f;   // 1/sqrt(64)
        float alpha[8];
#pragma unroll
        for (int v = 0; v < 8; ++v) {
            float a0 = s0[v] * sc, a1 = s1[v] * sc;
            s0[v] = a0; s1[v] = a1;
            float mx = fmaxf(a0, a1);
#pragma unroll
            for (int off = 1; off < 16; off <<= 1)
                mx = fmaxf(mx, __shfl_xor(mx, off, 32));
            float mn = fmaxf(mrow[v], mx);
            alpha[v] = __expf(mrow[v] - mn);
            mrow[v]  = mn;
        }
#pragma unroll
        for (int v = 0; v < 8; ++v) {
            float p0 = __expf(s0[v] - mrow[v]);
            float p1 = __expf(s1[v] - mrow[v]);
            float rs = p0 + p1;
#pragma unroll
            for (int off = 1; off < 16; off <<= 1)
                rs += __shfl_xor(rs, off, 32);
            lrow[v] = lrow[v] * alpha[v] + rs;
            // stage P (C-layout -> row-major 16x32 in LDS)
            const int m = v + 8 * hi;
            P[m * 32 + (l & 15)]      = (half_t)p0;
            P[m * 32 + 16 + (l & 15)] = (half_t)p1;
        }
#pragma unroll
        for (int v = 0; v < 8; ++v) {
            o0[v] *= alpha[v]; o1[v] *= alpha[v];
            o2[v] *= alpha[v]; o3[v] *= alpha[v];
        }
        asm volatile("s_wait_dscnt 0" ::: "memory");
        v16h aP = load_a_frag(P, 32);
        o0 = wmma16(aP, bV0, o0);
        o1 = wmma16(aP, bV1, o1);
        o2 = wmma16(aP, bV2, o2);
        o3 = wmma16(aP, bV3, o3);
    }

    // normalize + store as [B, T, H*DK] f16 for the output projection
    const int b = bh >> 4;
    const int h = bh & 15;
    v8f* oo[4] = {&o0, &o1, &o2, &o3};
#pragma unroll
    for (int j = 0; j < 4; ++j) {
        const int col = h * DK + j * 16 + (l & 15);
#pragma unroll
        for (int v = 0; v < 8; ++v) {
            const int t = q0 + v + 8 * hi;
            float val = (*oo[j])[v] / lrow[v];
            AO[((size_t)(b * SEQ_T + t)) * D_MODEL + col] = (half_t)val;
        }
    }
}

// ---------------------------------------------------------------------------
// Kernel 3: output projection out = AO @ Wo^T  (f32 result).
// grid = (32, 16), block = 128. Wave -> 32x64 tile.
// ---------------------------------------------------------------------------
__global__ __launch_bounds__(128) void out_proj_kernel(
    const half_t* __restrict__ AO, const half_t* __restrict__ Wo16,
    float* __restrict__ out)
{
    const int wid = threadIdx.x >> 5;
    const int l   = threadIdx.x & 31;
    const int hi  = l >> 4;
    const int m0  = (blockIdx.x * 4 + wid) * 32;
    const int n0  = blockIdx.y * 64;

    v8f acc[2][4] = {};
    mm32x64(AO, Wo16, m0, n0, acc);

#pragma unroll
    for (int mi = 0; mi < 2; ++mi)
#pragma unroll
        for (int j = 0; j < 4; ++j) {
            const int n = n0 + j * 16 + (l & 15);
#pragma unroll
            for (int v = 0; v < 8; ++v) {
                const int m = m0 + mi * 16 + v + 8 * hi;
                out[(size_t)m * D_MODEL + n] = acc[mi][j][v];
            }
        }
}

// ---------------------------------------------------------------------------
// Workspace layout (half_t elements):
//   [ 0M) q16  [ 4M) k16  [ 8M) v16
//   [12M) Wq16 [13M) Wk16 [14M) Wv16 [15M) Wo16
//   [16M) Qh   [20M) Kh   [24M) Vt   [28M) AO     (total 32M halves = 64 MB)
// ---------------------------------------------------------------------------
extern "C" void kernel_launch(void* const* d_in, const int* in_sizes, int n_in,
                              void* d_out, int out_size, void* d_ws, size_t ws_size,
                              hipStream_t stream) {
    (void)in_sizes; (void)n_in; (void)out_size; (void)ws_size;

    const float* q  = (const float*)d_in[0];
    const float* k  = (const float*)d_in[1];
    const float* vv = (const float*)d_in[2];
    const float* Wq = (const float*)d_in[3];
    const float* Wk = (const float*)d_in[4];
    const float* Wv = (const float*)d_in[5];
    const float* Wo = (const float*)d_in[6];

    half_t* ws = (half_t*)d_ws;
    const size_t MEG = 1024ull * 1024ull;
    half_t* q16  = ws + 0 * MEG;
    half_t* k16  = ws + 4 * MEG;
    half_t* v16  = ws + 8 * MEG;
    half_t* Wq16 = ws + 12 * MEG;
    half_t* Wk16 = ws + 13 * MEG;
    half_t* Wv16 = ws + 14 * MEG;
    half_t* Wo16 = ws + 15 * MEG;
    half_t* Qh   = ws + 16 * MEG;
    half_t* Kh   = ws + 20 * MEG;
    half_t* Vt   = ws + 24 * MEG;
    half_t* AO   = ws + 28 * MEG;

    const int nAct = M_TOT * D_MODEL;      // 4M
    const int nW   = D_MODEL * D_MODEL;    // 1M
    cvt_f32_f16_kernel<<<nAct / 256, 256, 0, stream>>>(q,  q16,  nAct);
    cvt_f32_f16_kernel<<<nAct / 256, 256, 0, stream>>>(k,  k16,  nAct);
    cvt_f32_f16_kernel<<<nAct / 256, 256, 0, stream>>>(vv, v16,  nAct);
    cvt_f32_f16_kernel<<<nW   / 256, 256, 0, stream>>>(Wq, Wq16, nW);
    cvt_f32_f16_kernel<<<nW   / 256, 256, 0, stream>>>(Wk, Wk16, nW);
    cvt_f32_f16_kernel<<<nW   / 256, 256, 0, stream>>>(Wv, Wv16, nW);
    cvt_f32_f16_kernel<<<nW   / 256, 256, 0, stream>>>(Wo, Wo16, nW);

    qkv_rope_kernel<<<dim3(32, NH, 3), 128, 0, stream>>>(
        q16, k16, v16, Wq16, Wk16, Wv16, Qh, Kh, Vt);

    attn_kernel<<<dim3(32, NB * NH), 128, 0, stream>>>(Qh, Kh, Vt, AO);

    out_proj_kernel<<<dim3(32, 16), 128, 0, stream>>>(AO, Wo16, (float*)d_out);
}
